// ModelClass_19327352832549
// MI455X (gfx1250) — compile-verified, matrix operand order checked by reading, not run on hardware
//
#include <hip/hip_runtime.h>

#define NLAYERS 4
#define NPER    65536
#define NTOT    (NLAYERS * NPER)   // 262144
#define DYN     48
#define STAT    16
#define NF      64
#define INIT_F  4
#define NGRAPH  128
#define E_UP    (4 * NTOT)         // 1048576
#define E_L     (8 * NPER)         // 524288

typedef __attribute__((ext_vector_type(16))) __bf16 v16bf;
typedef __attribute__((ext_vector_type(8)))  float  v8f;

// ---- f32 -> bf16 (round-to-nearest-even) bit helpers ----
__device__ __forceinline__ unsigned bfbits(float f) {
    union { float f; unsigned u; } v; v.f = f;
    unsigned u = v.u + 0x7FFFu + ((v.u >> 16) & 1u);
    return u >> 16;
}
__device__ __forceinline__ unsigned packbf(float lo, float hi) {
    return bfbits(lo) | (bfbits(hi) << 16);
}

// K index of element i (0..15) of a 16-bit WMMA A/B fragment, khalf = lane>>4.
// For i<8:  K = khalf*8 + i ; for i>=8: K = 16 + khalf*8 + (i-8)
__device__ __forceinline__ int kmap(int i, int khalf) {
    return ((i >> 3) << 4) + khalf * 8 + (i & 7);
}

// =====================================================================
// Weight pre-pack: W[K x n_out] (f32, K<=64 zero-padded) -> bf16 fragments.
// Layout: packed[ ((ct*2 + g)*32 + lane)*8 + p ]  (dword p = elems 2p,2p+1)
// so a wave loads B fragment (ct,g) as one contiguous 32B v16bf per lane.
// =====================================================================
__global__ void k_packW(const float* __restrict__ W, int K, int n_out,
                        unsigned* __restrict__ packed) {
    int idx = blockIdx.x * blockDim.x + threadIdx.x;
    int total = (n_out >> 4) * 2 * 32 * 8;
    if (idx >= total) return;
    int p    = idx & 7;
    int lane = (idx >> 3) & 31;
    int gct  = idx >> 8;
    int g    = gct & 1;
    int ct   = gct >> 1;
    int khalf = lane >> 4;
    int col   = ct * 16 + (lane & 15);
    int k0 = g * 32 + kmap(2 * p, khalf);
    int k1 = g * 32 + kmap(2 * p + 1, khalf);
    float f0 = (k0 < K) ? W[k0 * n_out + col] : 0.f;
    float f1 = (k1 < K) ? W[k1 * n_out + col] : 0.f;
    packed[idx] = packbf(f0, f1);
}

// Load 8 consecutive K-values starting at RB + khalf*8 from concat(A1,A2).
// All region decisions fold at compile time for the instantiations used.
template<int RB, int W1_, int W2_>
__device__ __forceinline__ void loadrun(const float* __restrict__ A1,
                                        const float* __restrict__ A2,
                                        long row, int khalf, float* buf) {
    constexpr int KTOT = W1_ + W2_;
    if constexpr (RB + 16 <= W1_) {
        const float* p = A1 + row * W1_ + RB + khalf * 8;
        *(float4*)(buf)     = *(const float4*)(p);
        *(float4*)(buf + 4) = *(const float4*)(p + 4);
    } else if constexpr (RB >= W1_ && RB + 16 <= KTOT) {
        const float* p = A2 + row * W2_ + (RB - W1_) + khalf * 8;
        *(float4*)(buf)     = *(const float4*)(p);
        *(float4*)(buf + 4) = *(const float4*)(p + 4);
    } else if constexpr (RB >= KTOT) {
#pragma unroll
        for (int t = 0; t < 8; ++t) buf[t] = 0.f;
    } else {
        // partial run (only the K=4 up-conv hits this, RB==0)
#pragma unroll
        for (int t = 0; t < 8; ++t) {
            int k = RB + khalf * 8 + t;
            float f = 0.f;
            if (k < W1_) f = A1[row * W1_ + k];
            else if constexpr (W2_ > 0) {
                if (k < KTOT) f = A2[row * W2_ + (k - W1_)];
            }
            buf[t] = f;
        }
    }
}

// =====================================================================
// OUT[m, NT*16] = act( concat(A1[m,W1_],A2[m,W2_]) @ Wpacked + bias )
// One wave per 16-row tile, 4 waves/block, m multiple of 64.
// =====================================================================
template<int W1_, int W2_, int NT, bool RIN, bool ROUT>
__global__ __launch_bounds__(128)
void wmma_gemm(const float* __restrict__ A1, const float* __restrict__ A2,
               const unsigned* __restrict__ Wp, const float* __restrict__ bias,
               float* __restrict__ OUT)
{
    constexpr int NOUT = NT * 16;
    const int lane  = threadIdx.x & 31;
    const int wave  = threadIdx.x >> 5;
    const int tile  = blockIdx.x * 4 + wave;
    const int khalf = lane >> 4;
    const long row  = (long)tile * 16 + (lane & 15);

    // ---- A: 4 contiguous 8-float runs -> two bf16 fragments ----
    float buf[32];
    loadrun< 0, W1_, W2_>(A1, A2, row, khalf, buf + 0);
    loadrun<16, W1_, W2_>(A1, A2, row, khalf, buf + 8);
    loadrun<32, W1_, W2_>(A1, A2, row, khalf, buf + 16);
    loadrun<48, W1_, W2_>(A1, A2, row, khalf, buf + 24);
    if constexpr (RIN) {
#pragma unroll
        for (int i = 0; i < 32; ++i) buf[i] = buf[i] > 0.f ? buf[i] : 0.f;
    }
    union { v16bf v; unsigned d[8]; } a0u, a1u;
#pragma unroll
    for (int p = 0; p < 8; ++p) {
        a0u.d[p] = packbf(buf[2 * p],      buf[2 * p + 1]);
        a1u.d[p] = packbf(buf[16 + 2 * p], buf[16 + 2 * p + 1]);
    }

    // ---- B fragments (pre-packed) + WMMA ----
    v8f acc[NT];
#pragma unroll
    for (int ct = 0; ct < NT; ++ct) {
        const v16bf b0 = *reinterpret_cast<const v16bf*>(Wp + ((ct * 2 + 0) * 32 + lane) * 8);
        const v16bf b1 = *reinterpret_cast<const v16bf*>(Wp + ((ct * 2 + 1) * 32 + lane) * 8);
        v8f c = {0.f, 0.f, 0.f, 0.f, 0.f, 0.f, 0.f, 0.f};
        c = __builtin_amdgcn_wmma_f32_16x16x32_bf16(false, a0u.v, false, b0,
                                                    (short)0, c, false, false);
        c = __builtin_amdgcn_wmma_f32_16x16x32_bf16(false, a1u.v, false, b1,
                                                    (short)0, c, false, false);
        acc[ct] = c;
    }

    // ---- writeback: C elem r -> (tile*16 + khalf*8 + r, ct*16 + lane&15) ----
    const long orow0 = (long)tile * 16 + khalf * 8;
#pragma unroll
    for (int ct = 0; ct < NT; ++ct) {
        int col = ct * 16 + (lane & 15);
        float bv = bias ? bias[col] : 0.f;
#pragma unroll
        for (int r = 0; r < 8; ++r) {
            float v = acc[ct][r] + bv;
            if constexpr (ROUT) v = v > 0.f ? v : 0.f;
            OUT[(orow0 + r) * NOUT + col] = v;
        }
    }
}

// ================= edge / aggregation kernels =================
__global__ void k_zero(float* p, long n) {
    long i = (long)blockIdx.x * blockDim.x + threadIdx.x;
    if (i < n) p[i] = 0.f;
}

__global__ void k_deg(const int* __restrict__ dst, int E, int doff,
                      float* __restrict__ deg) {
    int e = blockIdx.x * blockDim.x + threadIdx.x;
    if (e < E) atomicAdd(&deg[dst[e] + doff], 1.f);
}

__global__ void k_dinv(float* deg, int n) {
    int i = blockIdx.x * blockDim.x + threadIdx.x;
    if (i < n) deg[i] = rsqrtf(deg[i] + 1.f);
}

// out rows [row0, row0+nrows): out[r,:] = hw[r,:]*dinv[r]^2 + b
__global__ void k_selfinit(const float* __restrict__ hw, const float* __restrict__ dinv,
                           const float* __restrict__ b, float* __restrict__ out,
                           int row0, int nrows) {
    long idx = (long)blockIdx.x * blockDim.x + threadIdx.x;
    if (idx >= (long)nrows * DYN) return;
    int r = row0 + (int)(idx / DYN);
    int j = (int)(idx % DYN);
    float di = dinv[r];
    out[(long)r * DYN + j] = hw[(long)r * DYN + j] * di * di + b[j];
}

// out[dst,:] += hw[src,:] * dinv[src]*dinv[dst]
__global__ void k_scatter(const int* __restrict__ src, const int* __restrict__ dst,
                          int E, int soff, int doff,
                          const float* __restrict__ hw, const float* __restrict__ dinv,
                          float* __restrict__ out) {
    long idx = (long)blockIdx.x * blockDim.x + threadIdx.x;
    if (idx >= (long)E * 64) return;
    int e = (int)(idx >> 6);
    int j = (int)(idx & 63);
    if (j >= DYN) return;
    int s = src[e] + soff;
    int d = dst[e] + doff;
    float c = dinv[s] * dinv[d];
    atomicAdd(&out[(long)d * DYN + j], hw[(long)s * DYN + j] * c);
}

__global__ void k_pool(const float* __restrict__ h, const int* __restrict__ gid,
                       float* __restrict__ pooled) {
    long idx = (long)blockIdx.x * blockDim.x + threadIdx.x;
    if (idx >= (long)NTOT * DYN) return;
    int i = (int)(idx / DYN);
    int j = (int)(idx % DYN);
    atomicAdd(&pooled[gid[i] * DYN + j], h[idx]);
}

__global__ void k_head(const float* __restrict__ pooled, const float* __restrict__ Wend,
                       const float* __restrict__ bend, float* __restrict__ out) {
    int g = threadIdx.x;
    if (g >= NGRAPH) return;
    float a = bend[0];
#pragma unroll
    for (int k = 0; k < DYN; ++k) a += pooled[g * DYN + k] * Wend[k];
    out[g] = a > 0.f ? a : 0.f;
}

// =====================================================================
extern "C" void kernel_launch(void* const* d_in, const int* in_sizes, int n_in,
                              void* d_out, int out_size, void* d_ws, size_t ws_size,
                              hipStream_t stream)
{
    const float* x         = (const float*)d_in[0];
    const float* fs        = (const float*)d_in[1];
    const int*   up_src    = (const int*)d_in[2];
    const int*   up_dst    = (const int*)d_in[3];
    const int*   inner_src = (const int*)d_in[4];
    const int*   inner_dst = (const int*)d_in[5];
    const int*   fwd_src   = (const int*)d_in[6];
    const int*   fwd_dst   = (const int*)d_in[7];
    const int*   bwd_src   = (const int*)d_in[8];
    const int*   bwd_dst   = (const int*)d_in[9];
    const int*   gid       = (const int*)d_in[10];
    const float* W_up = (const float*)d_in[11]; const float* b_up = (const float*)d_in[12];
    const float* W_in = (const float*)d_in[13]; const float* b_in = (const float*)d_in[14];
    const float* W_f  = (const float*)d_in[15]; const float* b_f  = (const float*)d_in[16];
    const float* W_b  = (const float*)d_in[17]; const float* b_b  = (const float*)d_in[18];
    const float* W1   = (const float*)d_in[19]; const float* b1   = (const float*)d_in[20];
    const float* W2   = (const float*)d_in[21]; const float* b2   = (const float*)d_in[22];
    const float* W3   = (const float*)d_in[23]; const float* b3   = (const float*)d_in[24];
    const float* Wend = (const float*)d_in[25]; const float* bend = (const float*)d_in[26];
    float* out = (float*)d_out;
    (void)in_sizes; (void)n_in; (void)out_size; (void)ws_size;

    char* ws = (char*)d_ws;
    float*    h      = (float*)ws;    ws += (size_t)NTOT * DYN * 4;
    float*    hw     = (float*)ws;    ws += (size_t)NTOT * DYN * 4;
    float*    deg    = (float*)ws;    ws += (size_t)NTOT * 4;
    float*    tA     = (float*)ws;    ws += (size_t)NPER * 64 * 4;
    float*    tB     = (float*)ws;    ws += (size_t)NPER * 64 * 4;
    float*    pooled = (float*)ws;    ws += (size_t)NGRAPH * DYN * 4;
    unsigned* pW_up  = (unsigned*)ws; ws += 2048 * 4;
    unsigned* pW_in  = (unsigned*)ws; ws += 2048 * 4;
    unsigned* pW_f   = (unsigned*)ws; ws += 2048 * 4;
    unsigned* pW_b   = (unsigned*)ws; ws += 2048 * 4;
    unsigned* pW1    = (unsigned*)ws; ws += 2048 * 4;
    unsigned* pW2    = (unsigned*)ws; ws += 2048 * 4;
    unsigned* pW3    = (unsigned*)ws; ws += 2048 * 4;

    // ---- pack all weight matrices into WMMA B-fragment layout ----
    auto pack = [&](const float* W, int K, int n_out, unsigned* dst) {
        int total = (n_out >> 4) * 2 * 32 * 8;
        k_packW<<<dim3((total + 255) / 256), 256, 0, stream>>>(W, K, n_out, dst);
    };
    pack(W_up, INIT_F, DYN, pW_up);
    pack(W_in, NF, DYN, pW_in);
    pack(W_f,  NF, DYN, pW_f);
    pack(W_b,  NF, DYN, pW_b);
    pack(W1,   NF, NF,  pW1);
    pack(W2,   NF, NF,  pW2);
    pack(W3,   NF, DYN, pW3);

    // aggregation half of a gcn (GEMM into hw must already be queued)
    auto gcn_agg = [&](const float* b, const int* es, const int* ed, int E,
                       int soff, int doff, int nn, float* outp, int row0, int nrows) {
        k_zero<<<dim3((nn + 255) / 256), 256, 0, stream>>>(deg, nn);
        k_deg<<<dim3((E + 255) / 256), 256, 0, stream>>>(ed, E, doff, deg);
        k_dinv<<<dim3((nn + 255) / 256), 256, 0, stream>>>(deg, nn);
        long si = (long)nrows * DYN;
        k_selfinit<<<dim3((unsigned)((si + 255) / 256)), 256, 0, stream>>>(
            hw, deg, b, outp, row0, nrows);
        long sc = (long)E * 64;
        k_scatter<<<dim3((unsigned)((sc + 255) / 256)), 256, 0, stream>>>(
            es, ed, E, soff, doff, hw, deg, outp);
    };

    // ---- upscale conv on full graph ----
    wmma_gemm<INIT_F, 0, 3, false, false><<<dim3(NTOT / 64), 128, 0, stream>>>(
        x, nullptr, pW_up, nullptr, hw);
    gcn_agg(b_up, up_src, up_dst, E_UP, 0, 0, NTOT, h, 0, NTOT);

    // ---- forward sweep ----
    for (int l = 0; l < NLAYERS; ++l) {
        int s = l * NPER;
        wmma_gemm<DYN, STAT, 3, false, false><<<dim3(NPER / 64), 128, 0, stream>>>(
            h + (size_t)s * DYN, fs + (size_t)s * STAT, pW_in, nullptr, hw);
        gcn_agg(b_in, inner_src + l * E_L, inner_dst + l * E_L, E_L, 0, 0,
                NPER, h + (size_t)s * DYN, 0, NPER);
        if (l == NLAYERS - 1) continue;
        wmma_gemm<DYN, STAT, 3, false, false><<<dim3(2 * NPER / 64), 128, 0, stream>>>(
            h + (size_t)s * DYN, fs + (size_t)s * STAT, pW_f, nullptr, hw);
        gcn_agg(b_f, fwd_src + l * E_L, fwd_dst + l * E_L, E_L, 0, NPER,
                2 * NPER, h + (size_t)s * DYN, NPER, NPER);
        int p = s + NPER;   // dnn on freshly updated next-layer block
        wmma_gemm<DYN, STAT, 4, true, true><<<dim3(NPER / 64), 128, 0, stream>>>(
            h + (size_t)p * DYN, fs + (size_t)p * STAT, pW1, b1, tA);
        wmma_gemm<NF, 0, 4, false, true><<<dim3(NPER / 64), 128, 0, stream>>>(
            tA, nullptr, pW2, b2, tB);
        wmma_gemm<NF, 0, 3, false, true><<<dim3(NPER / 64), 128, 0, stream>>>(
            tB, nullptr, pW3, b3, h + (size_t)p * DYN);
    }

    // ---- backward sweep ----
    for (int l = NLAYERS - 1; l >= 1; --l) {
        int s = (l - 1) * NPER;
        wmma_gemm<DYN, STAT, 3, false, false><<<dim3(2 * NPER / 64), 128, 0, stream>>>(
            h + (size_t)s * DYN, fs + (size_t)s * STAT, pW_b, nullptr, hw);
        gcn_agg(b_b, bwd_src + (l - 1) * E_L, bwd_dst + (l - 1) * E_L, E_L, NPER, 0,
                2 * NPER, h + (size_t)s * DYN, 0, NPER);
        // pi = inner gcn -> tA (48-wide)
        wmma_gemm<DYN, STAT, 3, false, false><<<dim3(NPER / 64), 128, 0, stream>>>(
            h + (size_t)s * DYN, fs + (size_t)s * STAT, pW_in, nullptr, hw);
        gcn_agg(b_in, inner_src + (l - 1) * E_L, inner_dst + (l - 1) * E_L, E_L, 0, 0,
                NPER, tA, 0, NPER);
        // dnn(concat(pi, fs)) -> h block
        wmma_gemm<DYN, STAT, 4, true, true><<<dim3(NPER / 64), 128, 0, stream>>>(
            tA, fs + (size_t)s * STAT, pW1, b1, tB);
        wmma_gemm<NF, 0, 4, false, true><<<dim3(NPER / 64), 128, 0, stream>>>(
            tB, nullptr, pW2, b2, tA);
        wmma_gemm<NF, 0, 3, false, true><<<dim3(NPER / 64), 128, 0, stream>>>(
            tA, nullptr, pW3, b3, h + (size_t)s * DYN);
    }

    // ---- graph pooling + head ----
    k_zero<<<dim3((NGRAPH * DYN + 255) / 256), 256, 0, stream>>>(pooled, NGRAPH * DYN);
    long pn = (long)NTOT * DYN;
    k_pool<<<dim3((unsigned)((pn + 255) / 256)), 256, 0, stream>>>(h, gid, pooled);
    k_head<<<dim3(1), 128, 0, stream>>>(pooled, Wend, bend, out);
}
